// Attention_65429531787682
// MI455X (gfx1250) — matrix operand order repaired and checked by reading, not compiled, not run on hardware
//
#include <hip/hip_runtime.h>
#include <hip/hip_bf16.h>

typedef __bf16 bf16;
typedef __attribute__((ext_vector_type(16))) __bf16 v16bf;
typedef __attribute__((ext_vector_type(8)))  __bf16 v8bf;
typedef __attribute__((ext_vector_type(8)))  float  v8f;

// exact parameter type of the async-LDS builtins (from hipcc diagnostic):
// int __vector(4) in AS(1) / AS(3)
typedef int v4i_ __attribute__((vector_size(16)));
typedef __attribute__((address_space(1))) v4i_* g_v4i_p;
typedef __attribute__((address_space(3))) v4i_* l_v4i_p;

#define HQ   32      // heads
#define DH   64      // head dim
#define NSEQ 2048    // sequence length
#define DIM  2048    // model dim
#define BATCH 2
#define NQKV (3*DIM) // 6144
#define ROWS (BATCH*NSEQ) // 4096

#if __has_builtin(__builtin_amdgcn_global_load_async_to_lds_b128)
#define HAVE_ASYNC_LDS 1
#else
#define HAVE_ASYNC_LDS 0
#endif

// per-lane async global->LDS 16B copy (ASYNCcnt-tracked DMA on gfx1250)
static __device__ __forceinline__ void async_g2l_b128(const bf16* gsrc, bf16* ldst) {
#if HAVE_ASYNC_LDS
  __builtin_amdgcn_global_load_async_to_lds_b128(
      (g_v4i_p)(void*)gsrc, (l_v4i_p)(void*)ldst, 0, 0);
#else
  *(v8bf*)ldst = *(const v8bf*)gsrc;
#endif
}

static __device__ __forceinline__ void wait_async0() {
#if HAVE_ASYNC_LDS
#if __has_builtin(__builtin_amdgcn_s_wait_asynccnt)
  __builtin_amdgcn_s_wait_asynccnt(0);
#else
  asm volatile("s_wait_asynccnt 0x0" ::: "memory");
#endif
#endif
}

static __device__ __forceinline__ v16bf join8(v8bf lo, v8bf hi) {
  return __builtin_shufflevector(lo, hi, 0, 1, 2, 3, 4, 5, 6, 7,
                                 8, 9, 10, 11, 12, 13, 14, 15);
}

// ---------------------------------------------------------------- cast f32->bf16 (8 elems/thread)
__global__ __launch_bounds__(256) void cast_f32_bf16(const float* __restrict__ src,
                                                     bf16* __restrict__ dst, long long n) {
  long long i = ((long long)blockIdx.x * 256 + threadIdx.x) * 8;
  if (i + 7 < n) {
    float4 a = *(const float4*)(src + i);
    float4 b = *(const float4*)(src + i + 4);
    v8bf o;
    o[0] = (bf16)a.x; o[1] = (bf16)a.y; o[2] = (bf16)a.z; o[3] = (bf16)a.w;
    o[4] = (bf16)b.x; o[5] = (bf16)b.y; o[6] = (bf16)b.z; o[7] = (bf16)b.w;
    *(v8bf*)(dst + i) = o;
  }
}

// ---------------------------------------------------------------- bf16 WMMA GEMM
// C(MxN fp32) = A(MxK bf16 row-major) @ B(KxN bf16 row-major)
// Block tile 128x64, 256 threads = 8 waves; wave w owns rows w*16..w*16+15.
// Double-buffered LDS: async A-tile DMA + B-transpose for tile k+1 overlap WMMA on tile k.
#define BM 128
#define BN 64
#define BK 32
#define LDA (BK + 8)   // 40 bf16 = 80B rows (16B multiple)
#define LDB (BK + 8)

__global__ __launch_bounds__(256) void gemm_bf16_wmma(const bf16* __restrict__ A,
                                                      const bf16* __restrict__ B,
                                                      float* __restrict__ C,
                                                      int M, int N, int K) {
  __shared__ alignas(16) bf16 As[2][BM][LDA];   // row-major A tiles
  __shared__ alignas(16) bf16 Bt[2][BN][LDB];   // column-major B tiles: Bt[n][k]
  const int tid   = threadIdx.x;
  const int wave  = tid >> 5;
  const int lane  = tid & 31;
  const int lhalf = lane & 15;
  const int lhi   = lane >> 4;
  const int m0 = blockIdx.y * BM;
  const int n0 = blockIdx.x * BN;
  const int nt = K / BK;

  const int brow = tid >> 3;           // B-tile k row 0..31
  const int bcol = (tid & 7) * 8;      // B-tile n col base

  v8f acc[4];
#pragma unroll
  for (int t = 0; t < 4; ++t)
#pragma unroll
    for (int r = 0; r < 8; ++r) acc[t][r] = 0.0f;

  // ---- prologue: stage tile 0 into buffer 0
#pragma unroll
  for (int ch = tid; ch < (BM * BK) / 8; ch += 256) {
    int r = ch >> 2;                   // BK/8 = 4 chunks per row
    int c = (ch & 3) * 8;
    async_g2l_b128(&A[(size_t)(m0 + r) * K + c], &As[0][r][c]);
  }
  {
    v8bf bv = *(const v8bf*)&B[(size_t)brow * N + n0 + bcol];
#pragma unroll
    for (int j = 0; j < 8; ++j) Bt[0][bcol + j][brow] = bv[j];
  }
  wait_async0();
  __syncthreads();

  for (int kt = 0; kt < nt; ++kt) {
    const int cur = kt & 1, nxt = cur ^ 1;
    const bool have_next = (kt + 1 < nt);
    v8bf bv;
    if (have_next) {
      // issue next B global load + async A-tile DMA before computing on current tile
      bv = *(const v8bf*)&B[(size_t)((kt + 1) * BK + brow) * N + n0 + bcol];
#pragma unroll
      for (int ch = tid; ch < (BM * BK) / 8; ch += 256) {
        int r = ch >> 2;
        int c = (ch & 3) * 8;
        async_g2l_b128(&A[(size_t)(m0 + r) * K + (kt + 1) * BK + c], &As[nxt][r][c]);
      }
    }

    // ---- compute on current buffer
    v16bf afrag = join8(*(const v8bf*)&As[cur][wave * 16 + lhalf][lhi * 8],
                        *(const v8bf*)&As[cur][wave * 16 + lhalf][16 + lhi * 8]);
#pragma unroll
    for (int t = 0; t < 4; ++t) {
      // bfrag[j] = B[kt*BK + (j<8 ? lhi*8 + j : 16 + lhi*8 + j-8)][n0 + t*16 + lhalf]
      v16bf bfrag = join8(*(const v8bf*)&Bt[cur][t * 16 + lhalf][lhi * 8],
                          *(const v8bf*)&Bt[cur][t * 16 + lhalf][16 + lhi * 8]);
      acc[t] = __builtin_amdgcn_wmma_f32_16x16x32_bf16(
          false, afrag, false, bfrag, (short)0, acc[t], false, false);
    }

    if (have_next) {
      // scatter-transpose next B tile (load wait lands here, behind the WMMAs)
#pragma unroll
      for (int j = 0; j < 8; ++j) Bt[nxt][bcol + j][brow] = bv[j];
    }
    wait_async0();
    __syncthreads();
  }

#pragma unroll
  for (int t = 0; t < 4; ++t)
#pragma unroll
    for (int r = 0; r < 8; ++r) {
      int m = m0 + wave * 16 + r + lhi * 8;
      int n = n0 + t * 16 + lhalf;
      C[(size_t)m * N + n] = acc[t][r];
    }
}

// ---------------------------------------------------------------- QKV post: l2norm + rope
// One wave handles one (b,n,h): lane l owns d = {l, l+32}.
__global__ __launch_bounds__(256) void qkv_post(const float* __restrict__ qkv,
                                                const float* __restrict__ freqs,
                                                bf16* __restrict__ Qh,
                                                bf16* __restrict__ Kh,
                                                bf16* __restrict__ Vh) {
  const int wave = threadIdx.x >> 5;
  const int lane = threadIdx.x & 31;
  const int item = blockIdx.x * 8 + wave;        // ((b*NSEQ)+n)*HQ + h
  const int h    = item % HQ;
  const int row  = item / HQ;                    // b*NSEQ + n
  const int n    = row % NSEQ;
  const int b    = row / NSEQ;

  const float* base = qkv + (size_t)row * NQKV + h * DH;
  float q0 = base[lane],          q1 = base[lane + 32];
  float k0 = base[DIM + lane],    k1 = base[DIM + lane + 32];
  float v0 = base[2*DIM + lane],  v1 = base[2*DIM + lane + 32];

  // l2 normalize q and k over the 64-dim head
  float sq = q0 * q0 + q1 * q1;
  float sk = k0 * k0 + k1 * k1;
#pragma unroll
  for (int m = 1; m < 32; m <<= 1) {
    sq += __shfl_xor(sq, m, 32);
    sk += __shfl_xor(sk, m, 32);
  }
  float iq = rsqrtf(fmaxf(sq, 1e-24f));
  float ik = rsqrtf(fmaxf(sk, 1e-24f));
  q0 *= iq; q1 *= iq; k0 *= ik; k1 *= ik;

  // RoPE on first 32 dims (element index == lane); partner = lane ^ 16
  float fr = freqs[(size_t)n * 32 + lane];
  float c  = cosf(fr), s = sinf(fr);
  float qp = __shfl_xor(q0, 16, 32);
  float kp = __shfl_xor(k0, 16, 32);
  float sign = (lane < 16) ? -1.0f : 1.0f;
  q0 = q0 * c + sign * qp * s;
  k0 = k0 * c + sign * kp * s;

  // store [b][h][n][d]
  size_t ob = (((size_t)b * HQ + h) * NSEQ + n) * DH;
  Qh[ob + lane] = (bf16)q0;  Qh[ob + lane + 32] = (bf16)q1;
  Kh[ob + lane] = (bf16)k0;  Kh[ob + lane + 32] = (bf16)k1;
  Vh[ob + lane] = (bf16)v0;  Vh[ob + lane + 32] = (bf16)v1;
}

// ---------------------------------------------------------------- flash attention
// grid: (BATCH*HQ, NSEQ/128). Each of 8 waves owns a 16-row query strip.
// Ks row-major [key][d] (score B-frags run along d) via async DMA;
// Vs transposed to [d][key] (P@V B-frags run along key). All frag reads = ds_load_b128.
#define LDK (DH + 8)   // 72 bf16 = 144B rows (16B multiple)

__global__ __launch_bounds__(256) void flash_attn(const bf16* __restrict__ Qh,
                                                  const bf16* __restrict__ Kh,
                                                  const bf16* __restrict__ Vh,
                                                  bf16* __restrict__ attnOut) {
  const int bh   = blockIdx.x;          // b*HQ + h
  const int qblk = blockIdx.y;
  const int wave = threadIdx.x >> 5;
  const int lane = threadIdx.x & 31;
  const int lhalf = lane & 15;
  const int lhi   = lane >> 4;

  const size_t headBase = (size_t)bh * NSEQ * DH;
  const int qrow0 = qblk * 128 + wave * 16;

  __shared__ alignas(16) bf16 Ks[64][LDK];         // [key][d]
  __shared__ alignas(16) bf16 Vt[DH][64 + 8];      // [d][key]
  __shared__ alignas(16) bf16 Ps[8][16][LDK];      // per-wave P staging

  // Q fragment (16 rows x 64 k-dims -> two 16x32 chunks), b128 loads from global
  v16bf qf[2];
  const bf16* qptr = Qh + headBase + (size_t)(qrow0 + lhalf) * DH;
#pragma unroll
  for (int kk = 0; kk < 2; ++kk)
    qf[kk] = join8(*(const v8bf*)&qptr[kk * 32 + lhi * 8],
                   *(const v8bf*)&qptr[kk * 32 + 16 + lhi * 8]);

  float mrow[8], lrow[8];
  v8f out[4];
#pragma unroll
  for (int r = 0; r < 8; ++r) { mrow[r] = -1e30f; lrow[r] = 0.0f; }
#pragma unroll
  for (int t = 0; t < 4; ++t)
#pragma unroll
    for (int r = 0; r < 8; ++r) out[t][r] = 0.0f;

  for (int kb = 0; kb < NSEQ; kb += 64) {
    // K block: async global->LDS DMA; V block: issue both loads, then transpose-scatter
    v8bf vv[2];
#pragma unroll
    for (int u = 0; u < 2; ++u) {
      int ch  = threadIdx.x + u * 256;
      int key = ch >> 3;               // DH/8 = 8 chunks per row
      int c   = (ch & 7) * 8;
      async_g2l_b128(&Kh[headBase + (size_t)(kb + key) * DH + c], &Ks[key][c]);
      vv[u] = *(const v8bf*)&Vh[headBase + (size_t)(kb + key) * DH + c];
    }
#pragma unroll
    for (int u = 0; u < 2; ++u) {
      int ch  = threadIdx.x + u * 256;
      int key = ch >> 3;
      int c   = (ch & 7) * 8;
#pragma unroll
      for (int j = 0; j < 8; ++j) Vt[c + j][key] = vv[u][j];
    }
    if (kb + 64 < NSEQ) {
      __builtin_prefetch(&Kh[headBase + (size_t)(kb + 64 + (threadIdx.x >> 2)) * DH], 0, 1);
      __builtin_prefetch(&Vh[headBase + (size_t)(kb + 64 + (threadIdx.x >> 2)) * DH], 0, 1);
    }
    wait_async0();
    __syncthreads();

    // S(16x64) = Q(16x64) @ K^T(64x64)
    v8f s[4];
#pragma unroll
    for (int t = 0; t < 4; ++t)
#pragma unroll
      for (int r = 0; r < 8; ++r) s[t][r] = 0.0f;
#pragma unroll
    for (int t = 0; t < 4; ++t)
#pragma unroll
      for (int kk = 0; kk < 2; ++kk) {
        // bfrag[j] = K[t*16+lhalf][kk*32 + (j<8 ? lhi*8 + j : 16 + lhi*8 + j-8)]
        v16bf bfrag = join8(*(const v8bf*)&Ks[t * 16 + lhalf][kk * 32 + lhi * 8],
                            *(const v8bf*)&Ks[t * 16 + lhalf][kk * 32 + 16 + lhi * 8]);
        s[t] = __builtin_amdgcn_wmma_f32_16x16x32_bf16(
            false, qf[kk], false, bfrag, (short)0, s[t], false, false);
      }

    // scale + online softmax (row r of this lane-group = query row r + 8*lhi)
#pragma unroll
    for (int t = 0; t < 4; ++t)
#pragma unroll
      for (int r = 0; r < 8; ++r) s[t][r] *= 0.125f;

#pragma unroll
    for (int r = 0; r < 8; ++r) {
      float v = -1e30f;
#pragma unroll
      for (int t = 0; t < 4; ++t) v = fmaxf(v, s[t][r]);
#pragma unroll
      for (int msk = 1; msk < 16; msk <<= 1) v = fmaxf(v, __shfl_xor(v, msk, 32));
      float mnew  = fmaxf(mrow[r], v);
      float alpha = __expf(mrow[r] - mnew);
      mrow[r] = mnew;
      float rs = 0.0f;
#pragma unroll
      for (int t = 0; t < 4; ++t) {
        float p = __expf(s[t][r] - mnew);
        s[t][r] = p;
        rs += p;
      }
#pragma unroll
      for (int msk = 1; msk < 16; msk <<= 1) rs += __shfl_xor(rs, msk, 32);
      lrow[r] = lrow[r] * alpha + rs;
#pragma unroll
      for (int t = 0; t < 4; ++t) out[t][r] *= alpha;
    }

    // stage P (C-layout) into LDS, reload as A fragments (b128 reads)
#pragma unroll
    for (int t = 0; t < 4; ++t)
#pragma unroll
      for (int r = 0; r < 8; ++r)
        Ps[wave][r + 8 * lhi][t * 16 + lhalf] = (bf16)s[t][r];
    __syncthreads();

    // O += P(16x64) @ V(64x64)
#pragma unroll
    for (int kk = 0; kk < 2; ++kk) {
      v16bf pf = join8(*(const v8bf*)&Ps[wave][lhalf][kk * 32 + lhi * 8],
                       *(const v8bf*)&Ps[wave][lhalf][kk * 32 + 16 + lhi * 8]);
#pragma unroll
      for (int t = 0; t < 4; ++t) {
        // vfrag[j] = V[kk*32 + (j<8 ? lhi*8 + j : 16 + lhi*8 + j-8)][t*16+lhalf]
        v16bf vfrag = join8(*(const v8bf*)&Vt[t * 16 + lhalf][kk * 32 + lhi * 8],
                            *(const v8bf*)&Vt[t * 16 + lhalf][kk * 32 + 16 + lhi * 8]);
        out[t] = __builtin_amdgcn_wmma_f32_16x16x32_bf16(
            false, pf, false, vfrag, (short)0, out[t], false, false);
      }
    }
    __syncthreads();
  }

  // finalize and store bf16 into [b*NSEQ + n][h*64 + d] layout for the output GEMM
  const int b = bh >> 5;       // HQ == 32
  const int h = bh & 31;
#pragma unroll
  for (int t = 0; t < 4; ++t)
#pragma unroll
    for (int r = 0; r < 8; ++r) {
      int m   = qrow0 + r + 8 * lhi;
      int col = h * DH + t * 16 + lhalf;
      float o = out[t][r] / fmaxf(lrow[r], 1e-30f);
      attnOut[((size_t)b * NSEQ + m) * DIM + col] = (bf16)o;
    }
}

// ---------------------------------------------------------------- launcher
extern "C" void kernel_launch(void* const* d_in, const int* in_sizes, int n_in,
                              void* d_out, int out_size, void* d_ws, size_t ws_size,
                              hipStream_t stream) {
  const float* x     = (const float*)d_in[0];   // (2,2048,2048)
  const float* wqkv  = (const float*)d_in[1];   // (2048,6144)
  const float* wout  = (const float*)d_in[2];   // (2048,2048)
  const float* freqs = (const float*)d_in[3];   // (2048,32)
  float* outp        = (float*)d_out;

  const long long nX    = (long long)ROWS * DIM;     // 8M
  const long long nWqkv = (long long)DIM * NQKV;     // 12M
  const long long nWout = (long long)DIM * DIM;      // 4M
  const long long nHead = (long long)ROWS * DIM;     // 8M per Q/K/V/attnOut

  bf16* xb      = (bf16*)d_ws;
  bf16* wqkvb   = xb + nX;
  bf16* woutb   = wqkvb + nWqkv;
  bf16* Qh      = woutb + nWout;
  bf16* Kh      = Qh + nHead;
  bf16* Vh      = Kh + nHead;
  bf16* attnOut = Vh + nHead;
  float* qkv    = (float*)(attnOut + nHead);         // 4096 x 6144 fp32

  // 1) cast weights/activations to bf16 (vectorized, 8 elems/thread)
  cast_f32_bf16<<<(unsigned)(nX / 2048),    256, 0, stream>>>(x,    xb,    nX);
  cast_f32_bf16<<<(unsigned)(nWqkv / 2048), 256, 0, stream>>>(wqkv, wqkvb, nWqkv);
  cast_f32_bf16<<<(unsigned)(nWout / 2048), 256, 0, stream>>>(wout, woutb, nWout);

  // 2) QKV projection: (4096x2048) @ (2048x6144) -> fp32
  {
    dim3 grid(NQKV / BN, ROWS / BM);
    gemm_bf16_wmma<<<grid, 256, 0, stream>>>(xb, wqkvb, qkv, ROWS, NQKV, DIM);
  }

  // 3) l2norm + RoPE + head split
  qkv_post<<<(ROWS * HQ) / 8, 256, 0, stream>>>(qkv, freqs, Qh, Kh, Vh);

  // 4) flash attention per head
  {
    dim3 grid(BATCH * HQ, NSEQ / 128);
    flash_attn<<<grid, 256, 0, stream>>>(Qh, Kh, Vh, attnOut);
  }

  // 5) output projection: (4096x2048) @ (2048x2048) -> fp32 d_out
  {
    dim3 grid(DIM / BN, ROWS / BM);
    gemm_bf16_wmma<<<grid, 256, 0, stream>>>(attnOut, woutb, outp, ROWS, DIM, DIM);
  }
}
